// CGCN_27273042330405
// MI455X (gfx1250) — matrix operand order repaired and checked by reading, not compiled, not run on hardware
//
#include <hip/hip_runtime.h>
#include <hip/hip_bf16.h>

typedef __attribute__((ext_vector_type(16))) _Float16 v16h;
typedef __attribute__((ext_vector_type(8)))  float    v8f;

#define NUM_USER   50000
#define NUM_ITEM   25000
#define NUM_NODES  75000
#define DIMC       64
#define DIMF       1024
#define NEDGE      1000000
#define NITEM_T    16384

__device__ __forceinline__ float lrelu(float x, float s) { return x > 0.f ? x : s * x; }
// order-preserving float->uint encoding for atomicMax-based segment max
__device__ __forceinline__ unsigned fenc(float f) {
    unsigned u = __float_as_uint(f);
    return (u & 0x80000000u) ? ~u : (u | 0x80000000u);
}
__device__ __forceinline__ float fdec(unsigned u) {
    return (u & 0x80000000u) ? __uint_as_float(u & 0x7fffffffu) : __uint_as_float(~u);
}

// ---------------------------------------------------------------------------
// Pack f32 weight [K,64] into per-lane-contiguous f16 WMMA B-operand layout:
//   index = ((tileN*ksteps + t)*32 + lane)*16 halves
//   lane<16 : N = tile*16+lane,    elems j -> K = t*32 + j      (j=0..15)
//   lane>=16: N = tile*16+lane-16, elems j -> K = t*32 + 16 + j
// ---------------------------------------------------------------------------
__global__ void pack_b_f16(const float* __restrict__ W, _Float16* __restrict__ Bp, int K) {
    int ksteps = K >> 5;
    int total  = 4 * ksteps * 32;
    int id = blockIdx.x * blockDim.x + threadIdx.x;
    if (id >= total) return;
    int lane = id & 31;
    int t    = (id >> 5) % ksteps;
    int n    = (id / (32 * ksteps)) * 16 + (lane & 15);
    int kb   = t * 32 + ((lane >> 4) ? 16 : 0);
    v16h v;
#pragma unroll
    for (int j = 0; j < 16; ++j) v[j] = (_Float16)W[(size_t)(kb + j) * 64 + n];
    *(v16h*)(Bp + (size_t)id * 16) = v;
}

// ---------------------------------------------------------------------------
// C[M,64] = A[M,K] @ B (prepacked f16), f32 accumulate via v_wmma_f32_16x16x32_f16
// Block = 128 threads = 4 waves; wave w owns N-tile w; block owns 16 rows.
// mode 0: C = leaky_relu(acc + bias, 0.01)          (feature extractor)
// mode 1: C = acc (raw h); also s_src=h@avs, s_dst=h@avd via LDS reduction
// ---------------------------------------------------------------------------
__global__ void __launch_bounds__(128) gemm_wmma_n64(
    const float* __restrict__ A, const _Float16* __restrict__ Bp,
    const float* __restrict__ bias, float* __restrict__ C,
    const float* __restrict__ avs, const float* __restrict__ avd,
    float* __restrict__ ssrc, float* __restrict__ sdst,
    int M, int K, int mode)
{
    __shared__ float sh0[16];
    __shared__ float sh1[16];
    const int tid   = threadIdx.x;
    const int wave  = tid >> 5;
    const int lane  = tid & 31;
    const int hsel  = lane >> 4;
    const int l16   = lane & 15;
    const int row0  = blockIdx.x * 16;
    const int n     = wave * 16 + l16;
    const int ksteps = K >> 5;

    if (tid < 16) { sh0[tid] = 0.f; sh1[tid] = 0.f; }
    __syncthreads();

    int arow = row0 + l16; if (arow >= M) arow = M - 1;            // clamp loads
    const float*    ap = A + (size_t)arow * K + (hsel ? 16 : 0);
    const _Float16* bp = Bp + ((size_t)(wave * ksteps) * 32 + lane) * 16;

    v8f acc = {};
    for (int t = 0; t < ksteps; ++t) {
        const float* a0 = ap + t * 32;
        float4 x0 = *(const float4*)(a0);
        float4 x1 = *(const float4*)(a0 + 4);
        float4 x2 = *(const float4*)(a0 + 8);
        float4 x3 = *(const float4*)(a0 + 12);
        __builtin_prefetch(a0 + 64, 0, 3);                          // global_prefetch_b8
        v16h a;
        a[0] = (_Float16)x0.x; a[1] = (_Float16)x0.y; a[2] = (_Float16)x0.z; a[3] = (_Float16)x0.w;
        a[4] = (_Float16)x1.x; a[5] = (_Float16)x1.y; a[6] = (_Float16)x1.z; a[7] = (_Float16)x1.w;
        a[8]  = (_Float16)x2.x; a[9]  = (_Float16)x2.y; a[10] = (_Float16)x2.z; a[11] = (_Float16)x2.w;
        a[12] = (_Float16)x3.x; a[13] = (_Float16)x3.y; a[14] = (_Float16)x3.z; a[15] = (_Float16)x3.w;
        v16h b = *(const v16h*)(bp + (size_t)t * 32 * 16);
        acc = __builtin_amdgcn_wmma_f32_16x16x32_f16(false, a, false, b,
                                                     (short)0, acc, false, false);
    }

    float an = 0.f, ad = 0.f;
    if (mode == 1) { an = avs[n]; ad = avd[n]; }
    const float bn = bias ? bias[n] : 0.f;

#pragma unroll
    for (int i = 0; i < 8; ++i) {
        const int mi = i + (hsel ? 8 : 0);          // C/D layout: lanes16-31 hold M=i+8
        const int m  = row0 + mi;
        float v = acc[i];
        if (mode == 0) {
            v = lrelu(v + bn, 0.01f);
            if (m < M) C[(size_t)m * 64 + n] = v;
        } else {
            if (m < M) C[(size_t)m * 64 + n] = v;
            atomicAdd(&sh0[mi], v * an);
            atomicAdd(&sh1[mi], v * ad);
        }
    }
    __syncthreads();
    if (mode == 1 && tid < 16 && (row0 + tid) < M) {
        ssrc[row0 + tid] = sh0[tid];
        sdst[row0 + tid] = sh1[tid];
    }
}

// out[row] = normalize(base[row] (+ add[row])) ; one wave32 per 64-wide row
__global__ void __launch_bounds__(256) row_norm_add(
    const float* __restrict__ base, const float* __restrict__ add,
    float* __restrict__ out, int nrows)
{
    int row  = blockIdx.x * 8 + (threadIdx.x >> 5);
    int lane = threadIdx.x & 31;
    if (row >= nrows) return;
    size_t o = (size_t)row * 64;
    float v0 = base[o + lane], v1 = base[o + lane + 32];
    if (add) { v0 += add[o + lane]; v1 += add[o + lane + 32]; }
    float ss = v0 * v0 + v1 * v1;
#pragma unroll
    for (int m = 16; m >= 1; m >>= 1) ss += __shfl_xor(ss, m, 32);
    float inv = 1.f / fmaxf(sqrtf(ss), 1e-12f);
    out[o + lane] = v0 * inv; out[o + lane + 32] = v1 * inv;
}

__global__ void init_pass(float* __restrict__ outg, unsigned* __restrict__ menc,
                          float* __restrict__ den, const float* __restrict__ bias)
{
    int i = blockIdx.x * blockDim.x + threadIdx.x;
    if (i < NUM_NODES * 64) outg[i] = bias[i & 63];
    if (i < NUM_NODES) { menc[i] = 0u; den[i] = 0.f; }
}

__device__ __forceinline__ void edge_sd(const int* src, const int* dst, int E,
                                        int idx, int& s, int& d) {
    if (idx < E) { s = src[idx];     d = dst[idx]; }
    else         { s = dst[idx - E]; d = src[idx - E]; }   // symmetrized final pass
}

__global__ void edge_e_max(const int* __restrict__ src, const int* __restrict__ dst,
                           int E, int EE, const float* __restrict__ ssrc,
                           const float* __restrict__ sdst,
                           float* __restrict__ evals, unsigned* __restrict__ menc)
{
    int idx = blockIdx.x * blockDim.x + threadIdx.x;
    if (idx >= EE) return;
    int s, d; edge_sd(src, dst, E, idx, s, d);
    float e = lrelu(ssrc[s] + sdst[d], 0.2f);
    evals[idx] = e;
    atomicMax(&menc[d], fenc(e));
}

__global__ void edge_exp(const int* __restrict__ src, const int* __restrict__ dst,
                         int E, int EE, const float* __restrict__ evals,
                         const unsigned* __restrict__ menc,
                         float* __restrict__ exvals, float* __restrict__ den)
{
    int idx = blockIdx.x * blockDim.x + threadIdx.x;
    if (idx >= EE) return;
    int s, d; edge_sd(src, dst, E, idx, s, d);
    unsigned me = menc[d];
    float m = (me == 0u) ? 0.f : fdec(me);      // empty segment: -inf -> 0
    float ex = __expf(evals[idx] - m);
    exvals[idx] = ex;
    atomicAdd(&den[d], ex);
}

// one thread per (edge, col); 64 consecutive lanes hit one dst row (coalesced atomics)
__global__ void edge_agg(const int* __restrict__ src, const int* __restrict__ dst,
                         int E, int EE, const float* __restrict__ exvals,
                         const float* __restrict__ den, const float* __restrict__ h,
                         float* __restrict__ outg, float* __restrict__ alpha_out)
{
    int gi  = blockIdx.x * blockDim.x + threadIdx.x;
    int idx = gi >> 6, col = gi & 63;
    if (idx >= EE) return;
    int s, d; edge_sd(src, dst, E, idx, s, d);
    float al = exvals[idx] / (den[d] + 1e-16f);
    atomicAdd(&outg[(size_t)d * 64 + col], al * h[(size_t)s * 64 + col]);
    if (alpha_out && col == 0) alpha_out[idx] = al;
}

__global__ void finalize_out0(const float* __restrict__ x, const float* __restrict__ outg,
                              float* __restrict__ out)
{
    int i = blockIdx.x * blockDim.x + threadIdx.x;
    if (i < NUM_NODES * 64) out[i] = x[i] + lrelu(outg[i], 0.01f);
}

__global__ void write_zeros(float* p) { if (threadIdx.x < 3 && blockIdx.x == 0) p[threadIdx.x] = 0.f; }

__global__ void gather_tran(const int* __restrict__ items, const float* __restrict__ act,
                            float* __restrict__ out)
{
    int i = blockIdx.x * blockDim.x + threadIdx.x;
    if (i >= NITEM_T * 64) return;
    int r = items[i >> 6] - NUM_USER;
    out[i] = act[(size_t)r * 64 + (i & 63)];
}

extern "C" void kernel_launch(void* const* d_in, const int* in_sizes, int n_in,
                              void* d_out, int out_size, void* d_ws, size_t ws_size,
                              hipStream_t stream)
{
    const float* features_raw = (const float*)d_in[0];
    const float* preference   = (const float*)d_in[1];
    const float* Wg           = (const float*)d_in[2];
    const float* att_src      = (const float*)d_in[3];
    const float* att_dst      = (const float*)d_in[4];
    const float* bias         = (const float*)d_in[5];
    const float* student_W    = (const float*)d_in[6];
    const float* student_b    = (const float*)d_in[7];
    const int*   edge         = (const int*)d_in[8];
    const int*   items        = (const int*)d_in[9];
    const int*   src = edge;
    const int*   dst = edge + NEDGE;

    float* ws   = (float*)d_ws;
    float* X    = ws;                       // 75000*64
    float* H    = ws + 4800000;             // 75000*64
    float* OUTG = ws + 9600000;             // 75000*64
    float* ACT  = ws + 14400000;            // 25000*64 leaky(feats_lin)
    float* SSRC = ws + 16000000;            // 75008
    float* SDST = ws + 16075008;            // 75008
    unsigned* MENC = (unsigned*)(ws + 16150016); // 75008
    float* DEN  = ws + 16225024;            // 75008
    float* EV   = ws + 16300032;            // 2,000,000
    float* EX   = ws + 18300032;            // 2,000,000
    _Float16* BPF = (_Float16*)(ws + 20300032); // 65536 halves
    _Float16* BPW = (_Float16*)(ws + 20332800); // 4096 halves

    float* out0     = (float*)d_out;
    float* outAlpha = out0 + 4800000;
    float* outZ     = out0 + 6800000;
    float* outTran  = out0 + 6800003;

    // pack B operands once
    pack_b_f16<<<(4 * 32 * 32 + 255) / 256, 256, 0, stream>>>(student_W, BPF, DIMF);
    pack_b_f16<<<(4 * 2 * 32 + 255) / 256, 256, 0, stream>>>(Wg, BPW, DIMC);

    // feature extractor: ACT = leaky(features_raw @ student_W + b)
    gemm_wmma_n64<<<(NUM_ITEM + 15) / 16, 128, 0, stream>>>(
        features_raw, BPF, student_b, ACT,
        nullptr, nullptr, nullptr, nullptr, NUM_ITEM, DIMF, 0);

    // x = [normalize(pref); normalize(ACT)]
    row_norm_add<<<(NUM_USER + 7) / 8, 256, 0, stream>>>(preference, nullptr, X, NUM_USER);
    row_norm_add<<<(NUM_ITEM + 7) / 8, 256, 0, stream>>>(ACT, nullptr,
                                                         X + (size_t)NUM_USER * 64, NUM_ITEM);

    for (int p = 0; p < 4; ++p) {
        const int EE = (p < 3) ? NEDGE : 2 * NEDGE;
        // Item rows of x never change across passes -> their h / s_src / s_dst
        // (computed in pass 0) stay valid; passes 1..3 only refresh user rows.
        const int Mrows = (p == 0) ? NUM_NODES : NUM_USER;   // 50000 = 3125*16, no tail
        gemm_wmma_n64<<<(Mrows + 15) / 16, 128, 0, stream>>>(
            X, BPW, nullptr, H, att_src, att_dst, SSRC, SDST, Mrows, DIMC, 1);
        init_pass<<<(NUM_NODES * 64 + 255) / 256, 256, 0, stream>>>(OUTG, MENC, DEN, bias);
        edge_e_max<<<(EE + 255) / 256, 256, 0, stream>>>(src, dst, NEDGE, EE, SSRC, SDST, EV, MENC);
        edge_exp<<<(EE + 255) / 256, 256, 0, stream>>>(src, dst, NEDGE, EE, EV, MENC, EX, DEN);
        edge_agg<<<(EE * 64 + 255) / 256, 256, 0, stream>>>(
            src, dst, NEDGE, EE, EX, DEN, H, OUTG, (p == 3) ? outAlpha : nullptr);
        if (p < 3)
            row_norm_add<<<(NUM_USER + 7) / 8, 256, 0, stream>>>(X, OUTG, X, NUM_USER);
    }

    finalize_out0<<<(NUM_NODES * 64 + 255) / 256, 256, 0, stream>>>(X, OUTG, out0);
    write_zeros<<<1, 32, 0, stream>>>(outZ);
    gather_tran<<<(NITEM_T * 64 + 255) / 256, 256, 0, stream>>>(items, ACT, outTran);
}
